// BiquadFilter_45423574123131
// MI455X (gfx1250) — matrix operand order repaired and checked by reading, not compiled, not run on hardware
//
#include <hip/hip_runtime.h>
#include <hip/hip_bf16.h>
#include <math.h>

// ---------------------------------------------------------------------------
// Biquad-cascade -> frequency-sampled FIR -> causal FIR convolution.
//   k_H    : sample H[b,f] = prod_k Bf/Af on 4097 rfft bins         (tiny)
//   k_FIR  : fir[b,n] = irfft(H) via direct inverse rDFT            (~2 GFLOP)
//   k_XREV : reversed + f16 + zero-padded copy of input signal      (bandwidth)
//   k_BOP  : filter Hankel pre-swizzled into WMMA-B order, bank-tiled
//   k_CONV : 137 GFLOP conv on v_wmma_f32_16x16x32_f16, LDS-staged,
//            4-way M-tile register blocking (B operand reused 4x)
// ---------------------------------------------------------------------------

typedef _Float16 v8h  __attribute__((ext_vector_type(8)));
typedef _Float16 v16h __attribute__((ext_vector_type(16)));
typedef float    v8f  __attribute__((ext_vector_type(8)));

#define B_N     8
#define C_N     2
#define L_N     524288
#define K_N     6
#define FIRLEN  8192
#define NFFT    8192
#define F_N     (FIRLEN / 2 + 1)      // 4097 rfft bins
#define KCH     264                   // K-chunks of 32 (33 groups of 8), K_TOT=8448
#define GSZ     8                     // chunks per staged group
#define NGRP    33
#define XPAD    8448                  // zero pad: causal boundary + K overshoot
#define XRLEN   (L_N + XPAD)          // 532736
#define MT      4                     // M-tiles per wave (share B operand)
#define BLK_OUT 8192                  // outputs per block (8 waves * 4 tiles * 256)
#define XWIN    16640                 // staged window per block (halves)

#define TWO_PI_OVER_N 7.669903939428206e-4f  // 2*pi/8192

// --------------------------- stage 1: H[b,f] -------------------------------
__global__ void k_H(const float* __restrict__ Bs,
                    const float* __restrict__ A1p,
                    const float* __restrict__ A2p,
                    float2* __restrict__ H) {
  int f = blockIdx.x * blockDim.x + threadIdx.x;
  int b = blockIdx.y;
  if (f >= F_N) return;
  float w = TWO_PI_OVER_N * (float)f;
  float sw, cw;
  __sincosf(w, &sw, &cw);
  float c2w = cw * cw - sw * sw;
  float s2w = 2.0f * cw * sw;
  float hr = 1.0f, hi = 0.0f;
  for (int k = 0; k < K_N; ++k) {
    float b0 = Bs[(b * K_N + k) * 3 + 0];
    float b1 = Bs[(b * K_N + k) * 3 + 1];
    float b2 = Bs[(b * K_N + k) * 3 + 2];
    float a1 = 2.0f * tanhf(A1p[b * K_N + k]);
    float aa = fabsf(a1);
    float a2 = ((2.0f - aa) * tanhf(A2p[b * K_N + k]) + aa) * 0.5f;
    float nr = b0 + b1 * cw + b2 * c2w;
    float ni = -(b1 * sw + b2 * s2w);
    float dr = 1.0f + a1 * cw + a2 * c2w;
    float di = -(a1 * sw + a2 * s2w);
    float inv = 1.0f / (dr * dr + di * di);
    float qr = (nr * dr + ni * di) * inv;
    float qi = (ni * dr - nr * di) * inv;
    float t = hr * qr - hi * qi;
    hi = hr * qi + hi * qr;
    hr = t;
  }
  H[b * F_N + f] = make_float2(hr, hi);
}

// --------------------------- stage 2: fir = irfft(H) -----------------------
__global__ void k_FIR(const float2* __restrict__ H, float* __restrict__ fir) {
  int n = blockIdx.x * blockDim.x + threadIdx.x;  // 0..8191
  int b = blockIdx.y;
  float2 h0 = H[b * F_N + 0];
  float2 hN = H[b * F_N + (F_N - 1)];
  float edge = h0.x + ((n & 1) ? -hN.x : hN.x);
  float acc = 0.0f;
  for (int f = 1; f < F_N - 1; ++f) {
    float2 h = H[b * F_N + f];
    int pr = (f * n) & (NFFT - 1);          // exact angle reduction mod N
    float th = TWO_PI_OVER_N * (float)pr;
    float s, c;
    __sincosf(th, &s, &c);
    acc += h.x * c - h.y * s;
  }
  fir[b * FIRLEN + n] = (edge + 2.0f * acc) * (1.0f / (float)NFFT);
}

// ------------------- stage 3a: reversed padded f16 signal ------------------
__global__ void k_XREV(const float* __restrict__ x, _Float16* __restrict__ xrev) {
  int i = blockIdx.x * blockDim.x + threadIdx.x;
  int bc = blockIdx.y;
  if (i >= XRLEN) return;
  float v = (i < L_N) ? x[(size_t)bc * L_N + (L_N - 1 - i)] : 0.0f;
  xrev[(size_t)bc * XRLEN + i] = (_Float16)v;
}

// ------------- stage 3b: filter Hankel in WMMA-B order, bank-tiled ---------
// B[kg][j] = fir[kg + j - 15]; operand element (lane,e): col j=lane&15,
// K = 8*(lane>>4) + e (e<8), 16 + 8*(lane>>4) + e-8 (e>=8).  Chunk stored as 64 16B-units: unit(h,lane)=h*32+lane
// holding elements e=8h..8h+7, so each conv b128 load touches 32 consecutive
// units -> bank-conflict-free.
__global__ void k_BOP(const float* __restrict__ fir, _Float16* __restrict__ Bop) {
  int t = blockIdx.x * blockDim.x + threadIdx.x;  // (b, k0, lane)
  if (t >= B_N * KCH * 32) return;
  int lane = t & 31;
  int k0 = (t >> 5) % KCH;
  int b = t / (KCH * 32);
  int jj = lane & 15;
  int hi = lane >> 4;
  _Float16* chunk = Bop + (size_t)(b * KCH + k0) * 512;
  for (int h = 0; h < 2; ++h) {
    _Float16* dst = chunk + (h * 32 + lane) * 8;
    for (int e = 0; e < 8; ++e) {
      int fi = 32 * k0 + 8 * hi + 16 * h + e + jj - 15;
      float v = (fi >= 0 && fi < FIRLEN) ? fir[b * FIRLEN + fi] : 0.0f;
      dst[e] = (_Float16)v;
    }
  }
}

// --------------------------- stage 4: WMMA conv ----------------------------
// Block = 8 waves x 4 M-tiles = 8192 outputs of one (b,c).  Signal window
// (33 KB) staged once; B chunks double-buffered in groups of 8 with global
// loads issued before the 32-wmma burst.  Per chunk: 2 ds (B, reused by 4
// tiles) + 4x2 ds (A) feeding 4 wmma.  A loads hit 32 consecutive 16B LDS
// units across the wave (conflict-free); B layout tiled likewise.
__global__ void __launch_bounds__(256, 2)
k_CONV(const _Float16* __restrict__ xrev,
       const _Float16* __restrict__ Bop,
       float* __restrict__ out) {
  __shared__ _Float16 xs[XWIN];           // 33280 B signal window
  __shared__ _Float16 bsh[2][GSZ * 512];  // 2 x 8 KB B double buffer

  const int tid = threadIdx.x;
  const int lane = tid & 31;
  const int wave = tid >> 5;
  const int bc = blockIdx.y;
  const int b = bc >> 1;
  const int p_blk = blockIdx.x * BLK_OUT;
  const int m = lane & 15;
  const int hi = lane >> 4;

  // ---- stage signal window: xs[i] = xrev[bc][qbase + i]
  const size_t qbase = (size_t)bc * XRLEN + (size_t)(L_N - BLK_OUT - p_blk);
  for (int i = tid * 8; i < XWIN; i += 256 * 8)
    *(v8h*)(xs + i) = *(const v8h*)(xrev + qbase + i);

  // ---- stage B group 0
  const _Float16* bsrc = Bop + (size_t)b * KCH * 512;
  {
    v8h t0 = *(const v8h*)(bsrc + tid * 8);
    v8h t1 = *(const v8h*)(bsrc + (tid + 256) * 8);
    *(v8h*)(&bsh[0][0] + tid * 8) = t0;
    *(v8h*)(&bsh[0][0] + (tid + 256) * 8) = t1;
  }
  __syncthreads();

  // A row base inside xs for tile t, chunk kg: a_base - 256*t + kg (+16)
  const int a_base = 8176 - 1024 * wave - 16 * m + 8 * hi;

  v8f acc[MT];
#pragma unroll
  for (int t = 0; t < MT; ++t) acc[t] = (v8f){};

  for (int g = 0; g < NGRP; ++g) {
    const int cur = g & 1;
    // issue next group's global loads early (held in regs across compute)
    v8h r0 = {}, r1 = {};
    if (g + 1 < NGRP) {
      const _Float16* nsrc = bsrc + (size_t)(g + 1) * (GSZ * 512);
      r0 = *(const v8h*)(nsrc + tid * 8);
      r1 = *(const v8h*)(nsrc + (tid + 256) * 8);
    }
#pragma unroll
    for (int cc = 0; cc < GSZ; ++cc) {
      const int kg = (g * GSZ + cc) * 32;
      const _Float16* bb = &bsh[cur][cc * 512] + lane * 8;
      v8h b_lo = *(const v8h*)(bb);         // elements e=0..7
      v8h b_hi = *(const v8h*)(bb + 256);   // elements e=8..15
      v16h bv = __builtin_shufflevector(b_lo, b_hi,
          0, 1, 2, 3, 4, 5, 6, 7, 8, 9, 10, 11, 12, 13, 14, 15);
#pragma unroll
      for (int t = 0; t < MT; ++t) {
        const _Float16* ap = xs + (a_base - 256 * t + kg);
        v8h a_lo = *(const v8h*)(ap);       // K = kg + 8hi + 0..7
        v8h a_hi = *(const v8h*)(ap + 16);  // K = kg + 16 + 8hi + 0..7
        v16h a = __builtin_shufflevector(a_lo, a_hi,
            0, 1, 2, 3, 4, 5, 6, 7, 8, 9, 10, 11, 12, 13, 14, 15);
        acc[t] = __builtin_amdgcn_wmma_f32_16x16x32_f16(
            false, a, false, bv, (short)0, acc[t], false, false);
      }
    }
    __syncthreads();                      // everyone done reading bsh[cur]
    if (g + 1 < NGRP) {
      _Float16* dst = &bsh[cur ^ 1][0];
      *(v8h*)(dst + tid * 8) = r0;
      *(v8h*)(dst + (tid + 256) * 8) = r1;
    }
    __syncthreads();                      // bsh[next] visible to all waves
  }

  // C/D layout: lane<16 -> N=lane, M=v; lane>=16 -> N=lane-16, M=v+8.
  float* o = out + (size_t)bc * L_N + (p_blk + 1024 * wave) + (lane & 15);
#pragma unroll
  for (int t = 0; t < MT; ++t)
#pragma unroll
    for (int v = 0; v < 8; ++v)
      o[256 * t + 16 * (v + 8 * hi)] = acc[t][v];
}

// ---------------------------------------------------------------------------
extern "C" void kernel_launch(void* const* d_in, const int* in_sizes, int n_in,
                              void* d_out, int out_size, void* d_ws, size_t ws_size,
                              hipStream_t stream) {
  const float* x   = (const float*)d_in[0];   // [8,2,524288]
  const float* Bs  = (const float*)d_in[1];   // [8,6,3]
  const float* A1p = (const float*)d_in[2];   // [8,6]
  const float* A2p = (const float*)d_in[3];   // [8,6]
  float* out = (float*)d_out;                 // [8,2,524288]

  // workspace layout (offsets 128B-aligned), total ~19.8 MB
  char* ws = (char*)d_ws;
  float2*   H    = (float2*)(ws);              // 8*4097*8        =   262,208 B
  float*    fir  = (float*)(ws + 262400);      // 8*8192*4        =   262,144 B
  _Float16* Bop  = (_Float16*)(ws + 524544);   // 8*264*512*2     = 2,162,688 B
  _Float16* xrev = (_Float16*)(ws + 2687360);  // 16*532736*2     = 17,047,552 B
  (void)in_sizes; (void)n_in; (void)out_size; (void)ws_size;

  k_H   <<<dim3((F_N + 255) / 256, B_N),         256, 0, stream>>>(Bs, A1p, A2p, H);
  k_FIR <<<dim3(FIRLEN / 256, B_N),              256, 0, stream>>>(H, fir);
  k_XREV<<<dim3(XRLEN / 256, B_N * C_N),         256, 0, stream>>>(x, xrev);
  k_BOP <<<dim3((B_N * KCH * 32 + 255) / 256),   256, 0, stream>>>(fir, Bop);
  k_CONV<<<dim3(L_N / BLK_OUT, B_N * C_N),       256, 0, stream>>>(xrev, Bop, out);
}
